// FaceCapsNet_51075751084569
// MI455X (gfx1250) — matrix operand order repaired and checked by reading, not compiled, not run on hardware
//
#include <hip/hip_runtime.h>
#include <hip/hip_bf16.h>
#include <stdint.h>

// ---------------- types ----------------
typedef __attribute__((ext_vector_type(16))) __bf16 v16bf;
typedef __attribute__((ext_vector_type(8)))  float  v8f;

// ---------------- problem constants ----------------
#define BATCH   2
#define H_IN    250
#define C_IN    3
#define H1      242          // conv1 out spatial
#define C1_OUT  256
#define H2      117          // primarycaps out spatial
#define C2_OUT  128
#define NCAPS   219024       // 117*117*16
#define JCAPS   8

#define C1_M    (BATCH*H1*H1)        // 117128
#define C1_K    243                  // 9*9*3
#define C1_KS   8                    // K padded to 256, 8 steps of 32
#define C2_M    (BATCH*H2*H2)        // 27378
#define C2_K    20736                // 9*9*256
#define C2_KS   648                  // 20736/32

// Block tile: 128(M) x 128(N), 8 waves in 4x2, each wave 32(M) x 64(N)
#define TM      128
// LDS tile strides (padded to break bank conflicts; multiples of 8 bf16 for b128)
#define A_STR   40                   // 128 rows x 32 K
#define B_STR   136                  // 32 K rows x 128 N

// ---------------- WMMA fragment loaders (ISA 7.12.2 layouts) ----------------
// A 16x32 bf16: lane L -> M = L&15 ; K-halfgroup = L>>4
//   VGPR0..3 = K pairs {kb..kb+7}, VGPR4..7 = K pairs {kb+16..kb+23}, kb=(L>>4)*8
__device__ inline v16bf frag_a(const __bf16* Ald, int mbase, int lane) {
    int m  = mbase + (lane & 15);
    int kb = (lane >> 4) * 8;
    union { v16bf v; uint4 q[2]; } f;
    f.q[0] = *(const uint4*)(Ald + m * A_STR + kb);
    f.q[1] = *(const uint4*)(Ald + m * A_STR + kb + 16);
    return f.v;
}
// B 32x16 bf16: lane L -> N = L&15 ; halves 0..7 = K kb..kb+7, halves 8..15 = K kb+16..kb+23, kb=(L>>4)*8
__device__ inline v16bf frag_b(const __bf16* Bld, int nbase, int lane) {
    const __bf16* p = Bld + ((lane >> 4) * 8) * B_STR + nbase + (lane & 15);
    v16bf f;
#pragma unroll
    for (int i = 0; i < 8; ++i) {
        f[i]     = p[i * B_STR];
        f[i + 8] = p[(i + 16) * B_STR];
    }
    return f;
}

#define WMMA_BF16(A, B, C) \
    __builtin_amdgcn_wmma_f32_16x16x32_bf16(false, (A), false, (B), (short)0, (C), false, false)

// ---------------- weight fp32 -> bf16 ----------------
__global__ __launch_bounds__(256) void k_cvt_bf16(const float* __restrict__ s,
                                                  __bf16* __restrict__ d, long n) {
    long i = (long)blockIdx.x * 256 + threadIdx.x;
    if (i < n) d[i] = (__bf16)s[i];
}

// ---------------- Conv1: 9x9x3 -> 256, stride 1, bias + ReLU, out bf16 -------
// Implicit GEMM: M = B*242*242, N = 256 (tile 128 -> gridDim.y=2), K = 243 (pad 256)
// A-gather: each thread owns ONE K column (k = tid&31) and 16 M rows -> the
// (kh,kw,ci) decomposition happens once per K-step; guards are branchless.
__global__ __launch_bounds__(256) void k_conv1(const float* __restrict__ in,
                                               const __bf16* __restrict__ wA,
                                               const float* __restrict__ bias,
                                               __bf16* __restrict__ x1) {
    __shared__ __align__(16) __bf16 Ald[2][TM * A_STR];
    __shared__ __align__(16) __bf16 Bld[2][32 * B_STR];
    const int tid = threadIdx.x;
    const int mtile = blockIdx.x, ntile = blockIdx.y;

    // ---- A gather assignment: column k = tid&31, rows rgrp*16..+15 ----
    const int colk = tid & 31;
    const int rgrp = tid >> 5;               // 0..7
    int  aoff[16];                           // 32-bit row base offsets into `in`
    bool amval[16];
#pragma unroll
    for (int e = 0; e < 16; ++e) {
        const int  row = rgrp * 16 + e;
        const long gm  = (long)mtile * TM + row;
        amval[e] = gm < C1_M;
        const int g  = amval[e] ? (int)gm : 0;
        const int b  = g / (H1 * H1);
        const int r  = g % (H1 * H1);
        const int oy = r / H1, ox = r % H1;
        aoff[e] = ((b * H_IN + oy) * H_IN + ox) * C_IN;
    }

    // ---- B copy assignment: 8 threads per K row, 16 cols each ----
    const int bkr = tid >> 3;
    const int bchk = (tid & 7) * 16;

    const int lane = tid & 31;
    const int wave = tid >> 5;
    const int wm = (wave >> 1) * 32;         // wave row group (0..3)
    const int wn = (wave & 1) * 64;          // wave col group (0..1)
    v8f acc[2][4] = {};

    // staging registers (global -> regs -> LDS, split across the barrier)
    float av[16];
    uint4 bq0, bq1;

    auto gldA = [&](int ks) {
        const int k = ks * 32 + colk;
        const bool kval = k < C1_K;
        const int kk = kval ? k : 0;
        const int kh = kk / 27, rr = kk - kh * 27;
        const int kw = rr / 3,  ci = rr - kw * 3;
        const int off = (kh * H_IN + kw) * C_IN + ci;
#pragma unroll
        for (int e = 0; e < 16; ++e) {
            float v = in[aoff[e] + off];                  // always in-bounds
            av[e] = (kval && amval[e]) ? v : 0.f;         // branchless mask
        }
    };
    auto gldB = [&](int ks) {
        const int k = ks * 32 + bkr;
        uint4 z; z.x = z.y = z.z = z.w = 0u;
        bq0 = z; bq1 = z;
        if (k < C1_K) {
            const __bf16* p = wA + (long)k * C1_OUT + ntile * 128 + bchk;
            bq0 = *(const uint4*)(p);
            bq1 = *(const uint4*)(p + 8);
        }
    };
    auto stTile = [&](int buf) {
#pragma unroll
        for (int e = 0; e < 16; ++e)
            Ald[buf][(rgrp * 16 + e) * A_STR + colk] = (__bf16)av[e];
        *(uint4*)&Bld[buf][bkr * B_STR + bchk]     = bq0;
        *(uint4*)&Bld[buf][bkr * B_STR + bchk + 8] = bq1;
    };

    gldA(0); gldB(0);
    int buf = 0;
    for (int ks = 0; ks < C1_KS; ++ks) {
        stTile(buf);
        __syncthreads();
        if (ks + 1 < C1_KS) { gldA(ks + 1); gldB(ks + 1); }   // overlap with WMMA
        const v16bf a0 = frag_a(Ald[buf], wm,      lane);
        const v16bf a1 = frag_a(Ald[buf], wm + 16, lane);
        const v16bf b0 = frag_b(Bld[buf], wn,      lane);
        const v16bf b1 = frag_b(Bld[buf], wn + 16, lane);
        const v16bf b2 = frag_b(Bld[buf], wn + 32, lane);
        const v16bf b3 = frag_b(Bld[buf], wn + 48, lane);
        acc[0][0] = WMMA_BF16(a0, b0, acc[0][0]);
        acc[0][1] = WMMA_BF16(a0, b1, acc[0][1]);
        acc[0][2] = WMMA_BF16(a0, b2, acc[0][2]);
        acc[0][3] = WMMA_BF16(a0, b3, acc[0][3]);
        acc[1][0] = WMMA_BF16(a1, b0, acc[1][0]);
        acc[1][1] = WMMA_BF16(a1, b1, acc[1][1]);
        acc[1][2] = WMMA_BF16(a1, b2, acc[1][2]);
        acc[1][3] = WMMA_BF16(a1, b3, acc[1][3]);
        buf ^= 1;
    }
    // C store: lane L, vgpr r -> m = r + (L>>4)*8, n = L&15  (ISA 7.12.2)
#pragma unroll
    for (int sm = 0; sm < 2; ++sm)
#pragma unroll
        for (int sn = 0; sn < 4; ++sn)
#pragma unroll
            for (int r = 0; r < 8; ++r) {
                const long m = (long)mtile * TM + wm + sm * 16 + (lane >> 4) * 8 + r;
                const int  n = ntile * 128 + wn + sn * 16 + (lane & 15);
                if (m < C1_M) {
                    float v = acc[sm][sn][r] + bias[n];
                    v = v > 0.f ? v : 0.f;
                    x1[m * C1_OUT + n] = (__bf16)v;
                }
            }
}

// ---------------- Conv2 (PrimaryCaps): 9x9x256 -> 128, stride 2, bias, fp32 --
// M = B*117*117, N = 128 (one tile), K = 20736. Each K-chunk of 32 is a
// contiguous channel run inside one (kh,kw) tap -> pure vector loads.
__global__ __launch_bounds__(256) void k_conv2(const __bf16* __restrict__ x1,
                                               const __bf16* __restrict__ wB,
                                               const float* __restrict__ bias,
                                               float* __restrict__ pc) {
    __shared__ __align__(16) __bf16 Ald[2][TM * A_STR];
    __shared__ __align__(16) __bf16 Bld[2][32 * B_STR];
    const int tid = threadIdx.x;
    const int mtile = blockIdx.x;

    // A copy: 2 threads per row (128 rows), each 16 contiguous channels
    const int arow = tid >> 1;
    const int achk = (tid & 1) * 16;
    const long gm = (long)mtile * TM + arow;
    const bool mval = gm < C2_M;
    const long g = mval ? gm : 0;                 // clamp -> always-legal address
    const int b  = (int)(g / (H2 * H2));
    const int rr = (int)(g % (H2 * H2));
    const int oy = rr / H2, ox = rr % H2;
    const __bf16* inb = x1 + (((long)b * H1 + oy * 2) * H1 + ox * 2) * C1_OUT;

    const int bkr = tid >> 3;
    const int bchk = (tid & 7) * 16;

    const int lane = tid & 31;
    const int wave = tid >> 5;
    const int wm = (wave >> 1) * 32;
    const int wn = (wave & 1) * 64;
    v8f acc[2][4] = {};

    uint4 aq0, aq1, bq0, bq1;

    auto gldA = [&](int ks) {
        const int tap = ks >> 3;               // 256 channels = 8 chunks per tap
        const int cb  = (ks & 7) * 32;
        const int kh  = tap / 9, kw = tap - kh * 9;
        const __bf16* p = inb + ((long)kh * H1 + kw) * C1_OUT + cb + achk;
        aq0 = *(const uint4*)(p);
        aq1 = *(const uint4*)(p + 8);
        if (!mval) { aq0.x = aq0.y = aq0.z = aq0.w = 0u; aq1 = aq0; }
    };
    auto gldB = [&](int ks) {
        const __bf16* p = wB + ((long)ks * 32 + bkr) * C2_OUT + bchk;
        bq0 = *(const uint4*)(p);
        bq1 = *(const uint4*)(p + 8);
    };
    auto stTile = [&](int buf) {
        *(uint4*)&Ald[buf][arow * A_STR + achk]     = aq0;
        *(uint4*)&Ald[buf][arow * A_STR + achk + 8] = aq1;
        *(uint4*)&Bld[buf][bkr * B_STR + bchk]      = bq0;
        *(uint4*)&Bld[buf][bkr * B_STR + bchk + 8]  = bq1;
    };

    gldA(0); gldB(0);
    int buf = 0;
    for (int ks = 0; ks < C2_KS; ++ks) {
        stTile(buf);
        __syncthreads();
        if (ks + 1 < C2_KS) { gldA(ks + 1); gldB(ks + 1); }   // overlap with WMMA
        const v16bf a0 = frag_a(Ald[buf], wm,      lane);
        const v16bf a1 = frag_a(Ald[buf], wm + 16, lane);
        const v16bf b0 = frag_b(Bld[buf], wn,      lane);
        const v16bf b1 = frag_b(Bld[buf], wn + 16, lane);
        const v16bf b2 = frag_b(Bld[buf], wn + 32, lane);
        const v16bf b3 = frag_b(Bld[buf], wn + 48, lane);
        acc[0][0] = WMMA_BF16(a0, b0, acc[0][0]);
        acc[0][1] = WMMA_BF16(a0, b1, acc[0][1]);
        acc[0][2] = WMMA_BF16(a0, b2, acc[0][2]);
        acc[0][3] = WMMA_BF16(a0, b3, acc[0][3]);
        acc[1][0] = WMMA_BF16(a1, b0, acc[1][0]);
        acc[1][1] = WMMA_BF16(a1, b1, acc[1][1]);
        acc[1][2] = WMMA_BF16(a1, b2, acc[1][2]);
        acc[1][3] = WMMA_BF16(a1, b3, acc[1][3]);
        buf ^= 1;
    }
#pragma unroll
    for (int sm = 0; sm < 2; ++sm)
#pragma unroll
        for (int sn = 0; sn < 4; ++sn)
#pragma unroll
            for (int r = 0; r < 8; ++r) {
                const long m = (long)mtile * TM + wm + sm * 16 + (lane >> 4) * 8 + r;
                const int  n = wn + sn * 16 + (lane & 15);
                if (m < C2_M) pc[m * C2_OUT + n] = acc[sm][sn][r] + bias[n];
            }
}

// ---------------- squash primary caps: pc (flat groups of 8) -> u ------------
__global__ __launch_bounds__(256) void k_squash_u(const float* __restrict__ pc,
                                                  float* __restrict__ u) {
    const long g = (long)blockIdx.x * 256 + threadIdx.x;   // capsule index
    if (g >= (long)BATCH * NCAPS) return;
    const float* p = pc + g * 8;
    float t[8], sq = 0.f;
#pragma unroll
    for (int o = 0; o < 8; ++o) { t[o] = p[o]; sq += t[o] * t[o]; }
    const float sc = sq / (1.f + sq) / sqrtf(sq + 1e-7f);
    float* q = u + g * 8;
#pragma unroll
    for (int o = 0; o < 8; ++o) q[o] = t[o] * sc;
}

// ---------------- u_hat[b,j,i,o] = sum_d W[j,i,o,d] * u[b,i,d] ---------------
// one thread per (j,i,o); loops over batch so each 32B W row is read once.
__global__ __launch_bounds__(256) void k_uhat(const float* __restrict__ W,
                                              const float* __restrict__ u,
                                              float* __restrict__ uhat) {
    const long idx = (long)blockIdx.x * 256 + threadIdx.x;  // (j*N+i)*8+o
    if (idx >= (long)JCAPS * NCAPS * 8) return;
    const float* wr = W + idx * 8;
    float w[8];
#pragma unroll
    for (int d = 0; d < 8; ++d) w[d] = wr[d];
    const long i = (idx >> 3) % NCAPS;
#pragma unroll
    for (int b = 0; b < BATCH; ++b) {
        const float* uu = u + ((long)b * NCAPS + i) * 8;
        float s = 0.f;
#pragma unroll
        for (int d = 0; d < 8; ++d) s += w[d] * uu[d];
        uhat[(long)b * JCAPS * NCAPS * 8 + idx] = s;
    }
}

// ---------------- routing: s[b,j,:] = sum_i c * u_hat ------------------------
__global__ void k_zero_s(float* s) { if (threadIdx.x < BATCH * JCAPS * 8) s[threadIdx.x] = 0.f; }

__global__ __launch_bounds__(256) void k_routing_s(const float* __restrict__ uhat,
                                                   const float* __restrict__ blog,
                                                   float* __restrict__ s, int r) {
    const int bj = blockIdx.y;                 // b*8+j
    const int b = bj >> 3, j = bj & 7;
    float acc[8] = {0, 0, 0, 0, 0, 0, 0, 0};
    for (long i = (long)blockIdx.x * 256 + threadIdx.x; i < NCAPS; i += (long)gridDim.x * 256) {
        float c;
        if (r == 0) c = 1.0f / JCAPS;
        else {
            float bl[8], mx = -1e30f;
#pragma unroll
            for (int j2 = 0; j2 < 8; ++j2) { bl[j2] = blog[((long)b * 8 + j2) * NCAPS + i]; mx = fmaxf(mx, bl[j2]); }
            float sum = 0.f;
#pragma unroll
            for (int j2 = 0; j2 < 8; ++j2) sum += __expf(bl[j2] - mx);
            c = __expf(bl[j] - mx) / sum;
        }
        const float* up = uhat + ((long)bj * NCAPS + i) * 8;
#pragma unroll
        for (int o = 0; o < 8; ++o) acc[o] += c * up[o];
    }
    __shared__ float red[256 * 8];
#pragma unroll
    for (int o = 0; o < 8; ++o) red[threadIdx.x * 8 + o] = acc[o];
    __syncthreads();
    for (int st = 128; st > 0; st >>= 1) {
        if (threadIdx.x < st)
#pragma unroll
            for (int o = 0; o < 8; ++o) red[threadIdx.x * 8 + o] += red[(threadIdx.x + st) * 8 + o];
        __syncthreads();
    }
    if (threadIdx.x < 8) atomicAdd(&s[bj * 8 + threadIdx.x], red[threadIdx.x]);
}

// v = squash(s); always mirror into d_out (final iteration's value persists)
__global__ void k_squash_v(const float* __restrict__ s, float* __restrict__ v,
                           float* __restrict__ out) {
    const int t = threadIdx.x;
    if (t >= BATCH * JCAPS) return;
    float tv[8], sq = 0.f;
#pragma unroll
    for (int o = 0; o < 8; ++o) { tv[o] = s[t * 8 + o]; sq += tv[o] * tv[o]; }
    const float sc = sq / (1.f + sq) / sqrtf(sq + 1e-7f);
#pragma unroll
    for (int o = 0; o < 8; ++o) { const float x = tv[o] * sc; v[t * 8 + o] = x; out[t * 8 + o] = x; }
}

// b_logits update; overwrite at r==0 (workspace is poisoned, never memset)
__global__ __launch_bounds__(256) void k_update_b(const float* __restrict__ uhat,
                                                  const float* __restrict__ v,
                                                  float* __restrict__ blog, int r) {
    const long idx = (long)blockIdx.x * 256 + threadIdx.x;  // bj*N + i
    if (idx >= (long)BATCH * JCAPS * NCAPS) return;
    const long bj = idx / NCAPS;
    const float* up = uhat + idx * 8;
    const float* vp = v + bj * 8;
    float d = 0.f;
#pragma unroll
    for (int o = 0; o < 8; ++o) d += vp[o] * up[o];
    blog[idx] = (r == 0) ? d : blog[idx] + d;
}

// ---------------- host ----------------
extern "C" void kernel_launch(void* const* d_in, const int* in_sizes, int n_in,
                              void* d_out, int out_size, void* d_ws, size_t ws_size,
                              hipStream_t stream) {
    const float* in     = (const float*)d_in[0];
    const float* conv_w = (const float*)d_in[1];
    const float* conv_b = (const float*)d_in[2];
    const float* pc_w   = (const float*)d_in[3];
    const float* pc_b   = (const float*)d_in[4];
    const float* W_caps = (const float*)d_in[5];
    float* out = (float*)d_out;

    char* p = (char*)d_ws;
    auto carve = [&](size_t bytes) { void* q = (void*)p; p += (bytes + 255) & ~(size_t)255; return q; };
    const long n_w1 = 9L * 9 * 3 * 256;          // 62208
    const long n_w2 = 9L * 9 * 256 * 128;        // 2654208
    __bf16* wA   = (__bf16*)carve(n_w1 * 2);
    __bf16* wB   = (__bf16*)carve(n_w2 * 2);
    __bf16* x1   = (__bf16*)carve((long)C1_M * C1_OUT * 2);            // ~60 MB
    float*  pc   = (float*)carve((long)C2_M * C2_OUT * 4);             // ~14 MB
    float*  u    = (float*)carve((long)BATCH * NCAPS * 8 * 4);         // ~14 MB
    float*  uhat = (float*)carve((long)BATCH * JCAPS * NCAPS * 8 * 4); // ~112 MB
    float*  blog = (float*)carve((long)BATCH * JCAPS * NCAPS * 4);     // ~14 MB
    float*  s    = (float*)carve(BATCH * JCAPS * 8 * 4);
    float*  v    = (float*)carve(BATCH * JCAPS * 8 * 4);

    // 1) weights -> bf16
    k_cvt_bf16<<<(unsigned)((n_w1 + 255) / 256), 256, 0, stream>>>(conv_w, wA, n_w1);
    k_cvt_bf16<<<(unsigned)((n_w2 + 255) / 256), 256, 0, stream>>>(pc_w, wB, n_w2);

    // 2) conv1 (WMMA implicit GEMM)  grid: M-tiles x N-tiles
    k_conv1<<<dim3((C1_M + TM - 1) / TM, C1_OUT / 128), 256, 0, stream>>>(in, wA, conv_b, x1);

    // 3) primarycaps conv (WMMA implicit GEMM)
    k_conv2<<<dim3((C2_M + TM - 1) / TM, 1), 256, 0, stream>>>(x1, wB, pc_b, pc);

    // 4) squash -> u
    k_squash_u<<<(unsigned)(((long)BATCH * NCAPS + 255) / 256), 256, 0, stream>>>(pc, u);

    // 5) u_hat
    k_uhat<<<(unsigned)(((long)JCAPS * NCAPS * 8 + 255) / 256), 256, 0, stream>>>(W_caps, u, uhat);

    // 6) dynamic routing, 3 iterations
    for (int r = 0; r < 3; ++r) {
        k_zero_s<<<1, 128, 0, stream>>>(s);
        k_routing_s<<<dim3(128, BATCH * JCAPS), 256, 0, stream>>>(uhat, blog, s, r);
        k_squash_v<<<1, 32, 0, stream>>>(s, v, out);
        if (r < 2)
            k_update_b<<<(unsigned)(((long)BATCH * JCAPS * NCAPS + 255) / 256), 256, 0, stream>>>(uhat, v, blog, r);
    }
}
